// LayerLoss_37735582663030
// MI455X (gfx1250) — matrix operand order repaired and checked by reading, not compiled, not run on hardware
//
#include <hip/hip_runtime.h>
#include <hip/hip_bf16.h>
#include <math.h>

typedef __bf16 bf16_t;
typedef __attribute__((ext_vector_type(16))) __bf16 v16bf;
typedef __attribute__((ext_vector_type(8)))  __bf16 v8bf;
typedef __attribute__((ext_vector_type(8)))  float  v8f;
// exact pointee type the async-LDS builtin expects (from hipcc diagnostic)
typedef __attribute__((__vector_size__(4 * sizeof(int)))) int v4i_async;

#define AS1 __attribute__((address_space(1)))
#define AS3 __attribute__((address_space(3)))

#if defined(__gfx1250__) && \
    __has_builtin(__builtin_amdgcn_global_load_async_to_lds_b128) && \
    __has_builtin(__builtin_amdgcn_s_wait_asynccnt)
#define USE_ASYNC 1
#else
#define USE_ASYNC 0
#endif

namespace {
constexpr int   kB  = 8192;
constexpr int   kD  = 2048;
constexpr int   kN  = 16;
constexpr int   kM  = 128;
constexpr int   kO  = 16;
constexpr int   kNM = kN * kM;                      // 2048
constexpr float kInv = -1.0f / 2048.0f;             // -1/(2*sigma^2), sigma=32

constexpr int TILE = 128;   // C tile is TILE x TILE
constexpr int KC   = 64;    // K chunk staged in LDS
constexpr int LSTR = 72;    // LDS row stride (64 + 8 pad), keeps 16B alignment

// workspace layout (bytes)
constexpr size_t OFF_XBF  = 0;                                           // 32 MB bf16 x
constexpr size_t OFF_VT   = OFF_XBF + (size_t)kB  * kD * sizeof(bf16_t); // 8 MB bf16 V^T
constexpr size_t OFF_X2   = OFF_VT  + (size_t)kNM * kD * sizeof(bf16_t);
constexpr size_t OFF_V2   = OFF_X2  + (size_t)kB  * sizeof(float);
constexpr size_t OFF_KACC = OFF_V2  + (size_t)kNM * sizeof(float);       // 256 tile sums
constexpr size_t OFF_T2P  = OFF_KACC + 256  * sizeof(float);             // 1024 partials
constexpr size_t OFF_MSEP = OFF_T2P  + 1024 * sizeof(float);             // 128 partials
constexpr size_t OFF_SPP  = OFF_MSEP + 128  * sizeof(float);             // 128 partials
constexpr size_t OFF_GP   = OFF_SPP  + 128  * sizeof(float);             // 64*256 partials
} // namespace

// ---------- prep: x (f32) -> bf16 + row squared norms ----------
__global__ __launch_bounds__(256) void convert_x_kernel(
    const float* __restrict__ x, bf16_t* __restrict__ xb, float* __restrict__ x2) {
  __shared__ float red[256];
  const int row = blockIdx.x;
  const int t = threadIdx.x;
  const float* xr = x + (size_t)row * kD;
  bf16_t* xo = xb + (size_t)row * kD;
  float s = 0.f;
#pragma unroll
  for (int k = 0; k < kD / 256; ++k) {
    const int idx = k * 256 + t;
    const float v = xr[idx];
    s += v * v;
    xo[idx] = (bf16_t)v;
  }
  red[t] = s;
  __syncthreads();
  for (int w = 128; w > 0; w >>= 1) { if (t < w) red[t] += red[t + w]; __syncthreads(); }
  if (t == 0) x2[row] = red[0];
}

// ---------- prep: bases (N,D,M) -> Vt (N*M, D) bf16, via LDS transpose ----------
__global__ __launch_bounds__(256) void transpose_v_kernel(
    const float* __restrict__ bases, bf16_t* __restrict__ vt) {
  __shared__ float tile[64][65];
  const int bid = blockIdx.x;       // 16 n * 32 d-chunks * 2 m-chunks = 1024
  const int n   = bid >> 6;
  const int rem = bid & 63;
  const int d0  = (rem >> 1) * 64;
  const int m0  = (rem & 1) * 64;
  const int t = threadIdx.x;
#pragma unroll
  for (int i = 0; i < 16; ++i) {
    const int lin = i * 256 + t;
    const int dl = lin >> 6, ml = lin & 63;          // coalesced along m
    tile[dl][ml] = bases[((size_t)n * kD + (d0 + dl)) * kM + (m0 + ml)];
  }
  __syncthreads();
#pragma unroll
  for (int i = 0; i < 16; ++i) {
    const int lin = i * 256 + t;
    const int ml = lin >> 6, dl = lin & 63;          // coalesced along d
    vt[(size_t)(n * kM + m0 + ml) * kD + (d0 + dl)] = (bf16_t)tile[dl][ml];
  }
}

// ---------- prep: v2[n*M+m] = sum_d bases[n,d,m]^2 (coalesced along m) ----------
__global__ __launch_bounds__(128) void v2_kernel(
    const float* __restrict__ bases, float* __restrict__ v2) {
  const int n = blockIdx.x, m = threadIdx.x;
  float s = 0.f;
  for (int d = 0; d < kD; ++d) {
    const float v = bases[((size_t)n * kD + d) * kM + m];
    s += v * v;
  }
  v2[n * kM + m] = s;
}

// ---------- MSE + L1(betas) partials (same flat size: 131072) ----------
__global__ __launch_bounds__(256) void mse_sparse_kernel(
    const float* __restrict__ outv, const float* __restrict__ tgt,
    const float* __restrict__ betas, float* __restrict__ msep, float* __restrict__ spp) {
  __shared__ float r1[256], r2[256];
  const int t = threadIdx.x;
  const int total = kB * kO;
  float m = 0.f, s = 0.f;
  for (int idx = blockIdx.x * 256 + t; idx < total; idx += gridDim.x * 256) {
    const float d = outv[idx] - tgt[idx];
    m += d * d;
    s += fabsf(betas[idx]);
  }
  r1[t] = m; r2[t] = s;
  __syncthreads();
  for (int w = 128; w > 0; w >>= 1) {
    if (t < w) { r1[t] += r1[t + w]; r2[t] += r2[t + w]; }
    __syncthreads();
  }
  if (t == 0) { msep[blockIdx.x] = r1[0]; spp[blockIdx.x] = r2[0]; }
}

// ---------- G = beta^T beta partials (16x16), fixed-order grid stride ----------
__global__ __launch_bounds__(256) void beta_gram_kernel(
    const float* __restrict__ betas, float* __restrict__ gp) {
  __shared__ float row[16];
  const int t = threadIdx.x;
  const int i = t >> 4, j = t & 15;
  float p = 0.f;
  for (int b = blockIdx.x; b < kB; b += gridDim.x) {
    if (t < 16) row[t] = betas[(size_t)b * kN + t];
    __syncthreads();
    p += row[i] * row[j];
    __syncthreads();
  }
  gp[(size_t)blockIdx.x * 256 + t] = p;
}

// ---------- WMMA bf16 GEMM + fused exp epilogue ----------
// C[r,c] = sum_d A[r,d]*Bm[c,d]; epilogue: e = exp((rowN[r]+colN[c]-2C)*kInv)
// useBeta: weight e by betas[r*16 + bx] (each 128-col tile == one n block).
// outPart[blockIdx.x] = tile sum (deterministic tree reduce, single store).
// Async double-buffered pipeline: s_wait_asynccnt 0; barrier; issue next chunk
// global->LDS async; 16x wmma on current chunk.
__global__ __launch_bounds__(256) void gemm_exp_kernel(
    const bf16_t* __restrict__ A, const bf16_t* __restrict__ Bm,
    const float* __restrict__ rowN, const float* __restrict__ colN,
    const float* __restrict__ betas, float* __restrict__ outPart,
    int useBeta, int colTiles) {
#if USE_ASYNC
  __shared__ alignas(16) bf16_t As[2][TILE][LSTR];
  __shared__ alignas(16) bf16_t Bs[2][TILE][LSTR];
#else
  __shared__ alignas(16) bf16_t As[1][TILE][LSTR];
  __shared__ alignas(16) bf16_t Bs[1][TILE][LSTR];
#endif
  __shared__ float rowNS[TILE], colNS[TILE], betaS[TILE], red[256];

  const int t  = threadIdx.x;
  const int by = blockIdx.x / colTiles;
  const int bx = blockIdx.x % colTiles;
  const size_t rowBase = (size_t)by * TILE;
  const size_t colBase = (size_t)bx * TILE;
  const int wave = t >> 5;          // wave32
  const int lane = t & 31;
  const int r = lane & 15;          // row-in-frag (A) / column N (B,C)
  const int h = lane >> 4;          // lane half
  const int wy = wave >> 1;         // 0..3 -> row offset wy*32
  const int wx = wave & 1;          // 0..1 -> col offset wx*64

  v8f acc[2][4];
#pragma unroll
  for (int i = 0; i < 2; ++i)
#pragma unroll
    for (int jc = 0; jc < 4; ++jc)
#pragma unroll
      for (int e = 0; e < 8; ++e) acc[i][jc][e] = 0.f;

  // 16x wmma on LDS buffer `buf` at K-chunk base; fragments per ISA layouts.
  auto mma_chunk = [&](int buf) {
#pragma unroll
    for (int ks = 0; ks < KC; ks += 32) {
      // A frag (16x32 bf16): lane r, half h -> K = h*8+[0..8) and 16+h*8+[0..8)
      v16bf afr[2];
#pragma unroll
      for (int i = 0; i < 2; ++i) {
        const v8bf lo = *(const v8bf*)&As[buf][wy * 32 + i * 16 + r][ks + h * 8];
        const v8bf hi = *(const v8bf*)&As[buf][wy * 32 + i * 16 + r][ks + 16 + h * 8];
        afr[i] = __builtin_shufflevector(lo, hi, 0,1,2,3,4,5,6,7,8,9,10,11,12,13,14,15);
      }
      // B frag (32x16 bf16): lane column = r, K = h*8+[0..8) and 16+h*8+[0..8)
      v16bf bfr[4];
#pragma unroll
      for (int jc = 0; jc < 4; ++jc) {
        const v8bf b0 = *(const v8bf*)&Bs[buf][wx * 64 + jc * 16 + r][ks + h * 8];
        const v8bf b1 = *(const v8bf*)&Bs[buf][wx * 64 + jc * 16 + r][ks + 16 + h * 8];
        bfr[jc] = __builtin_shufflevector(b0, b1, 0,1,2,3,4,5,6,7,8,9,10,11,12,13,14,15);
      }
#pragma unroll
      for (int i = 0; i < 2; ++i)
#pragma unroll
        for (int jc = 0; jc < 4; ++jc)
          acc[i][jc] = __builtin_amdgcn_wmma_f32_16x16x32_bf16(
              false, afr[i], false, bfr[jc], (short)0, acc[i][jc], false, false);
    }
  };

#if USE_ASYNC
  // issue 128x64 bf16 tiles of A and B^T, global -> LDS, no VGPR round trip
  auto issue_chunk = [&](int buf, int k0) {
#pragma unroll
    for (int v = 0; v < 4; ++v) {
      const int lin = v * 256 + t;       // 1024 vectors of 8 bf16 per operand
      const int row = lin >> 3;
      const int kq  = (lin & 7) * 8;
      __builtin_amdgcn_global_load_async_to_lds_b128(
          (AS1 v4i_async*)(A + (rowBase + row) * kD + k0 + kq),
          (AS3 v4i_async*)&As[buf][row][kq], 0, 0);
      __builtin_amdgcn_global_load_async_to_lds_b128(
          (AS1 v4i_async*)(Bm + (colBase + row) * kD + k0 + kq),
          (AS3 v4i_async*)&Bs[buf][row][kq], 0, 0);
    }
  };

  constexpr int nK = kD / KC;
  issue_chunk(0, 0);                       // prologue: chunk 0 -> buffer 0
  for (int kc = 0; kc < nK; ++kc) {
    const int cur = kc & 1;
    __builtin_amdgcn_s_wait_asynccnt(0);   // this wave's async loads landed
    __syncthreads();                       // all waves' loads landed; prev buf free
    if (kc + 1 < nK) issue_chunk(cur ^ 1, (kc + 1) * KC);
    mma_chunk(cur);                        // overlaps next chunk's async flight
  }
#else
  for (int k0 = 0; k0 < kD; k0 += KC) {
    __syncthreads();
#pragma unroll
    for (int v = 0; v < 4; ++v) {
      const int lin = v * 256 + t;
      const int row = lin >> 3;
      const int kq  = (lin & 7) * 8;
      *(v8bf*)&As[0][row][kq] = *(const v8bf*)&A [(rowBase + row) * kD + k0 + kq];
      *(v8bf*)&Bs[0][row][kq] = *(const v8bf*)&Bm[(colBase + row) * kD + k0 + kq];
    }
    __syncthreads();
    mma_chunk(0);
  }
#endif

  __syncthreads();
  if (t < TILE) {
    rowNS[t] = rowN[rowBase + t];
    colNS[t] = colN[colBase + t];
    if (useBeta) betaS[t] = betas[(rowBase + t) * kN + bx];
    else         betaS[t] = 1.f;
  }
  __syncthreads();

  // fused epilogue: exp + weight + tile reduce
  float partial = 0.f;
#pragma unroll
  for (int i = 0; i < 2; ++i)
#pragma unroll
    for (int jc = 0; jc < 4; ++jc)
#pragma unroll
      for (int j = 0; j < 8; ++j) {
        // C layout: VGPR j, lane -> M = j + 8*h, N = lane&15
        const int rl = wy * 32 + i * 16 + j + 8 * h;
        const int cl = wx * 64 + jc * 16 + r;
        const float sq = rowNS[rl] + colNS[cl] - 2.f * acc[i][jc][j];
        partial += __expf(sq * kInv) * betaS[rl];
      }
  red[t] = partial;
  __syncthreads();
  for (int w = 128; w > 0; w >>= 1) { if (t < w) red[t] += red[t + w]; __syncthreads(); }
  if (t == 0) outPart[blockIdx.x] = red[0];
}

// ---------- finalize: combine partials, term3, power iteration, loss ----------
__global__ __launch_bounds__(256) void finalize_kernel(
    const float* __restrict__ kacc, const float* __restrict__ gp,
    const float* __restrict__ t2p,  const float* __restrict__ msep,
    const float* __restrict__ spp,  float* __restrict__ out) {
  __shared__ float red[256];
  __shared__ float K0s[16][16];
  __shared__ float vv[16], wv[16];
  __shared__ float sc[8]; // 0:mse 1:term2raw 2:sparse 3:term3raw 4:norm
  const int t = threadIdx.x;
  const int i = t >> 4, j = t & 15;

  float s = 0.f;
  for (int k = t; k < 1024; k += 256) s += t2p[k];
  red[t] = s; __syncthreads();
  for (int w = 128; w > 0; w >>= 1) { if (t < w) red[t] += red[t + w]; __syncthreads(); }
  if (t == 0) sc[1] = red[0];
  __syncthreads();

  red[t] = (t < 128) ? msep[t] : 0.f; __syncthreads();
  for (int w = 128; w > 0; w >>= 1) { if (t < w) red[t] += red[t + w]; __syncthreads(); }
  if (t == 0) sc[0] = red[0];
  __syncthreads();

  red[t] = (t < 128) ? spp[t] : 0.f; __syncthreads();
  for (int w = 128; w > 0; w >>= 1) { if (t < w) red[t] += red[t + w]; __syncthreads(); }
  if (t == 0) sc[2] = red[0];
  __syncthreads();

  // K = kacc / (M*M); G summed over 64 block partials; term3raw = sum K.*G
  float g = 0.f;
  for (int w = 0; w < 64; ++w) g += gp[w * 256 + t];
  const float kij = kacc[t] * (1.0f / ((float)kM * (float)kM));
  K0s[i][j] = (i == j) ? 0.f : kij;
  red[t] = kij * g; __syncthreads();
  for (int w = 128; w > 0; w >>= 1) { if (t < w) red[t] += red[t + w]; __syncthreads(); }
  if (t == 0) sc[3] = red[0];
  if (t < 16) vv[t] = 1.0f + 0.0625f * (float)t;  // deterministic start
  __syncthreads();

  // power iteration on symmetric K0: sigma_max = max|eig| = ||K0 v|| at fixpoint
  for (int it = 0; it < 128; ++it) {
    if (t < 16) {
      float a = 0.f;
      for (int c = 0; c < 16; ++c) a += K0s[t][c] * vv[c];
      wv[t] = a;
    }
    __syncthreads();
    if (t == 0) {
      float nn = 0.f;
      for (int c = 0; c < 16; ++c) nn += wv[c] * wv[c];
      sc[4] = sqrtf(nn);
    }
    __syncthreads();
    const float nm = sc[4];
    if (t < 16) vv[t] = (nm > 1e-30f) ? wv[t] / nm : 0.f;
    __syncthreads();
  }

  if (t == 0) {
    const float base   = sc[0] / (float)(kB * kO);
    const float term2  = sc[1] * (-2.0f / ((float)kB * (float)kM));
    const float term3  = sc[3] / (float)kB;
    const float sparse = sc[2] / (float)kB;
    const float sigma  = sc[4];
    // LAM_OLS = LAM_ORTH = LAM_SPARSE = 1, term1 = 1
    out[0] = base + (1.0f + term2 + term3) + sigma + sparse;
  }
}

extern "C" void kernel_launch(void* const* d_in, const int* in_sizes, int n_in,
                              void* d_out, int out_size, void* d_ws, size_t ws_size,
                              hipStream_t stream) {
  (void)in_sizes; (void)n_in; (void)out_size; (void)ws_size;
  const float* d_output = (const float*)d_in[0];
  const float* d_target = (const float*)d_in[1];
  const float* d_x      = (const float*)d_in[2];
  const float* d_bases  = (const float*)d_in[3];
  const float* d_betas  = (const float*)d_in[4];

  char* ws = (char*)d_ws;
  bf16_t* xbf  = (bf16_t*)(ws + OFF_XBF);
  bf16_t* vt   = (bf16_t*)(ws + OFF_VT);
  float*  x2   = (float*)(ws + OFF_X2);
  float*  v2   = (float*)(ws + OFF_V2);
  float*  Kacc = (float*)(ws + OFF_KACC);
  float*  t2p  = (float*)(ws + OFF_T2P);
  float*  msep = (float*)(ws + OFF_MSEP);
  float*  spp  = (float*)(ws + OFF_SPP);
  float*  gp   = (float*)(ws + OFF_GP);

  convert_x_kernel  <<<kB,   256, 0, stream>>>(d_x, xbf, x2);
  transpose_v_kernel<<<1024, 256, 0, stream>>>(d_bases, vt);
  v2_kernel         <<<kN,   128, 0, stream>>>(d_bases, v2);
  mse_sparse_kernel <<<128,  256, 0, stream>>>(d_output, d_target, d_betas, msep, spp);
  beta_gram_kernel  <<<64,   256, 0, stream>>>(d_betas, gp);

  // GEMM1: (8192 x 2048) x (2048 x 2048): 64 row tiles * 16 col tiles
  gemm_exp_kernel<<<(kB / TILE) * (kNM / TILE), 256, 0, stream>>>(
      xbf, vt, x2, v2, d_betas, t2p, /*useBeta=*/1, kNM / TILE);
  // GEMM2 (gram): (2048 x 2048) x (2048 x 2048): 16x16 tiles, one per K[n1,n2]
  gemm_exp_kernel<<<(kNM / TILE) * (kNM / TILE), 256, 0, stream>>>(
      vt, vt, v2, v2, d_betas, Kacc, /*useBeta=*/0, kNM / TILE);

  finalize_kernel<<<1, 256, 0, stream>>>(Kacc, gp, t2p, msep, spp, (float*)d_out);
}
